// SymbolicLightModel_22127671509482
// MI455X (gfx1250) — matrix operand order repaired and checked by reading, not compiled, not run on hardware
//
#include <hip/hip_runtime.h>

#define BB 2
#define SS 512
#define DIM 1536
#define NH 24
#define HD 64
#define FF 6144
#define VV 57344
#define NL 4
#define WINDOW 256
#define ANCHORS 4

typedef __attribute__((ext_vector_type(16))) __bf16 v16bf;
typedef __attribute__((ext_vector_type(2)))  __bf16 v2bf;
typedef __attribute__((ext_vector_type(8)))  float  v8f;
typedef __attribute__((ext_vector_type(2)))  float  v2f;

union Frag { v16bf v; uint4 u[2]; };

// Hardware f32 -> bf16 (RNE). Vector convert lowers to a single
// v_cvt_pk_bf16_f32 producing the packed dword directly.
__device__ inline unsigned short f2bf1(float f) {
  union { __bf16 b; unsigned short u; } r;
  r.b = (__bf16)f;
  return r.u;
}
__device__ inline unsigned int pack_bf2(float lo, float hi) {
  union { v2bf b; unsigned int u; } r;
  v2f s = {lo, hi};
  r.b = __builtin_convertvector(s, v2bf);
  return r.u;
}

// Async global->LDS copy of 32 bytes (two B128 ops; IOFFSET advances BOTH the
// global and the LDS address per CDNA5 ISA sec.10 async addressing).
__device__ inline void async_copy32(unsigned lds_off, const void* gsrc) {
  unsigned long long g = (unsigned long long)(size_t)gsrc;
  asm volatile("global_load_async_to_lds_b128 %0, %1, off\n\t"
               "global_load_async_to_lds_b128 %0, %1, off offset:16"
               :: "v"(lds_off), "v"(g) : "memory");
}
__device__ inline void wait_async0() {
#if __has_builtin(__builtin_amdgcn_s_wait_asynccnt)
  __builtin_amdgcn_s_wait_asynccnt(0);
#else
  asm volatile("s_wait_asynccnt 0" ::: "memory");
#endif
}

// ---------------------------------------------------------------------------
// Weight convert + transpose: W [K,N] f32  ->  WT [N,K] bf16
// ---------------------------------------------------------------------------
__global__ __launch_bounds__(256) void slm_wconv(const float* __restrict__ W,
                                                 unsigned short* __restrict__ WT,
                                                 int K, int N) {
  __shared__ unsigned short tile[32][33];
  const int nt = blockIdx.x * 32;
  const int kt = blockIdx.y * 32;
  const int x = threadIdx.x;   // 0..31
  const int y0 = threadIdx.y;  // 0..7
#pragma unroll
  for (int r = 0; r < 4; ++r) {
    int y = y0 + r * 8;
    tile[y][x] = f2bf1(W[(size_t)(kt + y) * N + nt + x]);
  }
  __syncthreads();
#pragma unroll
  for (int r = 0; r < 4; ++r) {
    int y = y0 + r * 8;
    WT[(size_t)(nt + y) * K + kt + x] = tile[x][y];
  }
}

// ---------------------------------------------------------------------------
// LayerNorm (optionally fused with embedding gather)
// ---------------------------------------------------------------------------
__global__ __launch_bounds__(256) void slm_ln(const int* __restrict__ ids,
                                              const float* __restrict__ embed,
                                              const float* __restrict__ src,
                                              const float* __restrict__ g,
                                              const float* __restrict__ b,
                                              float* __restrict__ out) {
  __shared__ float red[256];
  __shared__ float stats[2];
  const int tok = blockIdx.x;
  const int t = threadIdx.x;
  const float* row = ids ? (embed + (size_t)ids[tok] * DIM)
                         : (src + (size_t)tok * DIM);
  float s = 0.f, s2 = 0.f;
  for (int j = t; j < DIM; j += 256) { float v = row[j]; s += v; s2 += v * v; }
  red[t] = s; __syncthreads();
  for (int o = 128; o > 0; o >>= 1) { if (t < o) red[t] += red[t + o]; __syncthreads(); }
  if (t == 0) stats[0] = red[0] / (float)DIM;
  __syncthreads();
  red[t] = s2; __syncthreads();
  for (int o = 128; o > 0; o >>= 1) { if (t < o) red[t] += red[t + o]; __syncthreads(); }
  if (t == 0) {
    float m = stats[0];
    float var = red[0] / (float)DIM - m * m;
    stats[1] = rsqrtf(var + 1e-5f);
  }
  __syncthreads();
  float m = stats[0], rs = stats[1];
  for (int j = t; j < DIM; j += 256)
    out[(size_t)tok * DIM + j] = (row[j] - m) * rs * g[j] + b[j];
}

// ---------------------------------------------------------------------------
// WMMA GEMM: C[M,N] = A[M,K](f32) @ WT[N,K](bf16)^T  (+ scale*addend)
// 256 threads / 8 waves, 128x128 tile, K-step 32, double-buffered LDS.
// B tile: async global->LDS (bf16, no VGPR roundtrip, ASYNCcnt tracked).
// A tile: f32 prefetched to regs, v_cvt_pk_bf16_f32, ds_store (pipelined).
// ---------------------------------------------------------------------------
__global__ __launch_bounds__(256) void slm_gemm(const float* __restrict__ A,
                                                const unsigned short* __restrict__ WT,
                                                float* __restrict__ C,
                                                const float* __restrict__ addend,
                                                const float* __restrict__ scale_ptr,
                                                int M, int N, int K) {
  __shared__ unsigned short As[2][128 * 56];
  __shared__ unsigned short Bs[2][128 * 56];
  const int t = threadIdx.x;
  const int lane = t & 31;
  const int wid = t >> 5;
  const int wm = wid & 3;   // 4 waves along M (32 rows each)
  const int wn = wid >> 2;  // 2 waves along N (64 cols each)
  const int mBase = blockIdx.y * 128;
  const int nBase = blockIdx.x * 128;

  const v8f zero = {0.f, 0.f, 0.f, 0.f, 0.f, 0.f, 0.f, 0.f};
  v8f acc[2][4];
#pragma unroll
  for (int a = 0; a < 2; ++a)
#pragma unroll
    for (int c = 0; c < 4; ++c) acc[a][c] = zero;

  const int rowL = t >> 1;  // 0..127 staged row
  const int half = t & 1;   // which 16-element half of the K=32 slab
  const int koff = (lane & 16) >> 1;  // A frag: 0 or 8 elements
  const int band = (lane & 16) >> 1;  // B frag: 0 or 8 elements
  const int r16 = lane & 15;

  const float* aSrc = A + (size_t)(mBase + rowL) * K + half * 16;
  const unsigned short* bSrc = WT + (size_t)(nBase + rowL) * K + half * 16;
  const unsigned slot = (unsigned)(rowL * 56 + half * 16);
  const unsigned bLds0 = (unsigned)(size_t)&Bs[0][slot];
  const unsigned bufBytes = 128u * 56u * 2u;

  float4 af4[4];
  // ---- prologue: stage tile 0 into buffer 0 ----
  async_copy32(bLds0, bSrc);
  {
    const float4* s = (const float4*)aSrc;
#pragma unroll
    for (int q4 = 0; q4 < 4; ++q4) af4[q4] = s[q4];
    unsigned int* dst = (unsigned int*)&As[0][slot];
#pragma unroll
    for (int q4 = 0; q4 < 4; ++q4) {
      dst[q4 * 2 + 0] = pack_bf2(af4[q4].x, af4[q4].y);
      dst[q4 * 2 + 1] = pack_bf2(af4[q4].z, af4[q4].w);
    }
  }
  wait_async0();
  __syncthreads();

  const int nk = K >> 5;
  for (int kt = 0; kt < nk; ++kt) {
    const int cur = kt & 1;
    const int nxt = cur ^ 1;
    const bool has_next = (kt + 1) < nk;
    const int kb_next = (kt + 1) << 5;

    if (has_next) {
      async_copy32(bLds0 + (unsigned)nxt * bufBytes, bSrc + kb_next);
      const float4* s = (const float4*)(aSrc + kb_next);
#pragma unroll
      for (int q4 = 0; q4 < 4; ++q4) af4[q4] = s[q4];
    }

    // ---- compute from current buffer ----
    Frag af[2], bf[4];
#pragma unroll
    for (int im = 0; im < 2; ++im) {
      const unsigned short* base = &As[cur][(wm * 32 + im * 16 + r16) * 56];
      af[im].u[0] = *(const uint4*)(base + koff);        // K = koff..koff+7
      af[im].u[1] = *(const uint4*)(base + 16 + koff);   // K = 16+koff..23+koff
    }
#pragma unroll
    for (int in = 0; in < 4; ++in) {
      const unsigned short* base = &Bs[cur][(wn * 64 + in * 16 + r16) * 56 + band];
      bf[in].u[0] = *(const uint4*)(base);      // K = band..band+7
      bf[in].u[1] = *(const uint4*)(base + 16); // K = 16+band..23+band
    }
#pragma unroll
    for (int im = 0; im < 2; ++im)
#pragma unroll
      for (int in = 0; in < 4; ++in)
        acc[im][in] = __builtin_amdgcn_wmma_f32_16x16x32_bf16(
            false, af[im].v, false, bf[in].v, (short)0, acc[im][in], false, false);

    if (has_next) {
      unsigned int* dst = (unsigned int*)&As[nxt][slot];
#pragma unroll
      for (int q4 = 0; q4 < 4; ++q4) {
        dst[q4 * 2 + 0] = pack_bf2(af4[q4].x, af4[q4].y);
        dst[q4 * 2 + 1] = pack_bf2(af4[q4].z, af4[q4].w);
      }
    }
    wait_async0();
    __syncthreads();
  }

  const float sc = scale_ptr ? *scale_ptr : 1.0f;
  const int rAdd = (lane & 16) >> 1;  // +8 rows for high half-wave
#pragma unroll
  for (int im = 0; im < 2; ++im) {
#pragma unroll
    for (int in = 0; in < 4; ++in) {
      const int row0 = mBase + wm * 32 + im * 16 + rAdd;
      const int col = nBase + wn * 64 + in * 16 + r16;
#pragma unroll
      for (int r = 0; r < 8; ++r) {
        size_t idx = (size_t)(row0 + r) * N + col;
        float cv = acc[im][in][r];
        if (addend) cv += sc * addend[idx];
        C[idx] = cv;
      }
    }
  }
}

// ---------------------------------------------------------------------------
// RoPE in-place on [B,S,H,HD]
// ---------------------------------------------------------------------------
__global__ __launch_bounds__(256) void slm_rope(float* __restrict__ qk) {
  const int t = blockIdx.x * blockDim.x + threadIdx.x;
  if (t >= BB * SS * NH * 32) return;
  const int pair = t & 31;
  const int bsh = t >> 5;
  const int pos = (bsh / NH) % SS;
  float inv = __powf(10000.0f, -((float)(2 * pair)) / 64.0f);
  float ang = (float)pos * inv;
  float sn, cs;
  __sincosf(ang, &sn, &cs);
  float* p = qk + (size_t)bsh * HD;
  float x1 = p[pair], x2 = p[pair + 32];
  p[pair] = x1 * cs - x2 * sn;
  p[pair + 32] = x2 * cs + x1 * sn;
}

// ---------------------------------------------------------------------------
// Sliding-window + anchor attention. One block per (b,h,i) query row.
// ---------------------------------------------------------------------------
__global__ __launch_bounds__(256) void slm_attn(const float* __restrict__ q,
                                                const float* __restrict__ k,
                                                const float* __restrict__ v,
                                                float* __restrict__ o) {
  __shared__ float qs[HD];
  __shared__ float scb[WINDOW + ANCHORS];
  __shared__ float red[256];
  __shared__ float stats[2];
  const int i = blockIdx.x, h = blockIdx.y, b = blockIdx.z;
  const int t = threadIdx.x;
  int wsrt = i - (WINDOW - 1); if (wsrt < 0) wsrt = 0;
  int na = wsrt < ANCHORS ? wsrt : ANCHORS;
  const int ns = i - wsrt + 1;
  const int count = na + ns;

  if (t < HD) qs[t] = q[(size_t)(b * SS + i) * DIM + h * HD + t];
  __syncthreads();

  float lmax = -1e30f;
  for (int idx = t; idx < count; idx += 256) {
    int j = (idx < na) ? idx : (wsrt + idx - na);
    const float* krow = k + (size_t)(b * SS + j) * DIM + h * HD;
    float s = 0.f;
#pragma unroll 8
    for (int d = 0; d < HD; ++d) s += qs[d] * krow[d];
    s *= 0.125f;  // 1/sqrt(64)
    scb[idx] = s;
    lmax = fmaxf(lmax, s);
  }
  red[t] = lmax; __syncthreads();
  for (int off = 128; off > 0; off >>= 1) { if (t < off) red[t] = fmaxf(red[t], red[t + off]); __syncthreads(); }
  if (t == 0) stats[0] = red[0];
  __syncthreads();
  const float mx = stats[0];
  float lsum = 0.f;
  for (int idx = t; idx < count; idx += 256) {
    float w = __expf(scb[idx] - mx);
    scb[idx] = w;
    lsum += w;
  }
  red[t] = lsum; __syncthreads();
  for (int off = 128; off > 0; off >>= 1) { if (t < off) red[t] += red[t + off]; __syncthreads(); }
  if (t == 0) stats[1] = red[0];
  __syncthreads();
  const float inv = 1.0f / stats[1];
  if (t < HD) {
    float acc = 0.f;
    for (int idx = 0; idx < count; ++idx) {
      int j = (idx < na) ? idx : (wsrt + idx - na);
      acc += scb[idx] * v[(size_t)(b * SS + j) * DIM + h * HD + t];
    }
    o[(size_t)(b * SS + i) * DIM + h * HD + t] = acc * inv;
  }
}

// ---------------------------------------------------------------------------
// LIF scan over time: one thread per (b,f) channel.
// ---------------------------------------------------------------------------
__global__ __launch_bounds__(256) void slm_lif(const float* __restrict__ hid,
                                               float* __restrict__ spk) {
  const int t = blockIdx.x * blockDim.x + threadIdx.x;
  if (t >= BB * FF) return;
  const int b = t / FF, f = t % FF;
  const float* src = hid + (size_t)b * SS * FF + f;
  float* dst = spk + (size_t)b * SS * FF + f;
  float vm = 0.0f;
  for (int s = 0; s < SS; ++s) {
    vm = fminf(3.0f, fmaxf(-3.0f, vm * 0.95f + src[(size_t)s * FF]));
    float sp = (vm >= 1.0f) ? 1.0f : 0.0f;
    dst[(size_t)s * FF] = sp;
    vm *= (1.0f - sp);
  }
}

// ---------------------------------------------------------------------------
// Exact GELU in-place
// ---------------------------------------------------------------------------
__global__ __launch_bounds__(256) void slm_gelu(float* __restrict__ x, int n) {
  const int t = blockIdx.x * blockDim.x + threadIdx.x;
  if (t < n) {
    float v = x[t];
    x[t] = 0.5f * v * (1.0f + erff(v * 0.70710678118f));
  }
}

// ---------------------------------------------------------------------------
extern "C" void kernel_launch(void* const* d_in, const int* in_sizes, int n_in,
                              void* d_out, int out_size, void* d_ws, size_t ws_size,
                              hipStream_t stream) {
  const int* token_ids = (const int*)d_in[0];
  const float* embed = (const float*)d_in[1];
  const float* enc_g = (const float*)d_in[2];
  const float* enc_b = (const float*)d_in[3];
  const float* ln1_g = (const float*)d_in[4];
  const float* ln1_b = (const float*)d_in[5];
  const float* Wq = (const float*)d_in[6];
  const float* Wk = (const float*)d_in[7];
  const float* Wv = (const float*)d_in[8];
  const float* Wo = (const float*)d_in[9];
  const float* ln2_g = (const float*)d_in[10];
  const float* ln2_b = (const float*)d_in[11];
  const float* W1 = (const float*)d_in[12];
  const float* W2 = (const float*)d_in[13];
  const float* Wout = (const float*)d_in[14];
  const float* Wp1 = (const float*)d_in[15];
  const float* Wp2 = (const float*)d_in[16];
  const float* pw = (const float*)d_in[17];
  float* out = (float*)d_out;

  char* p = (char*)d_ws;
  auto alloc = [&](size_t bytes) -> char* {
    char* r = p;
    p += (bytes + 255) & ~(size_t)255;
    return r;
  };
  const size_t DD = (size_t)DIM * DIM, DF = (size_t)DIM * FF;
  unsigned short* wqT = (unsigned short*)alloc(NL * DD * 2);
  unsigned short* wkT = (unsigned short*)alloc(NL * DD * 2);
  unsigned short* wvT = (unsigned short*)alloc(NL * DD * 2);
  unsigned short* woT = (unsigned short*)alloc(NL * DD * 2);
  unsigned short* w1T = (unsigned short*)alloc(NL * DF * 2);
  unsigned short* w2T = (unsigned short*)alloc(NL * DF * 2);
  unsigned short* woutT = (unsigned short*)alloc((size_t)VV * DIM * 2);
  unsigned short* wp1T = (unsigned short*)alloc((size_t)384 * DIM * 2);
  unsigned short* wp2T = (unsigned short*)alloc((size_t)VV * 384 * 2);
  const size_t TOK = (size_t)BB * SS;
  float* x = (float*)alloc(TOK * DIM * 4);
  float* h = (float*)alloc(TOK * DIM * 4);
  float* qb = (float*)alloc(TOK * DIM * 4);
  float* kb = (float*)alloc(TOK * DIM * 4);
  float* vb = (float*)alloc(TOK * DIM * 4);
  float* ob = (float*)alloc(TOK * DIM * 4);
  float* hid = (float*)alloc(TOK * FF * 4);
  float* spk = (float*)alloc(TOK * FF * 4);
  float* p1 = (float*)alloc(TOK * 384 * 4);

  // ---- weight convert + transpose to bf16 [N,K] ----
  dim3 tb(32, 8);
  for (int l = 0; l < NL; ++l) {
    slm_wconv<<<dim3(DIM / 32, DIM / 32), tb, 0, stream>>>(Wq + l * DD, wqT + l * DD, DIM, DIM);
    slm_wconv<<<dim3(DIM / 32, DIM / 32), tb, 0, stream>>>(Wk + l * DD, wkT + l * DD, DIM, DIM);
    slm_wconv<<<dim3(DIM / 32, DIM / 32), tb, 0, stream>>>(Wv + l * DD, wvT + l * DD, DIM, DIM);
    slm_wconv<<<dim3(DIM / 32, DIM / 32), tb, 0, stream>>>(Wo + l * DD, woT + l * DD, DIM, DIM);
    slm_wconv<<<dim3(FF / 32, DIM / 32), tb, 0, stream>>>(W1 + l * DF, w1T + l * DF, DIM, FF);
    slm_wconv<<<dim3(DIM / 32, FF / 32), tb, 0, stream>>>(W2 + l * DF, w2T + l * DF, FF, DIM);
  }
  slm_wconv<<<dim3(VV / 32, DIM / 32), tb, 0, stream>>>(Wout, woutT, DIM, VV);
  slm_wconv<<<dim3(384 / 32, DIM / 32), tb, 0, stream>>>(Wp1, wp1T, DIM, 384);
  slm_wconv<<<dim3(VV / 32, 384 / 32), tb, 0, stream>>>(Wp2, wp2T, 384, VV);

  const int M = (int)TOK;  // 1024
  dim3 gD(DIM / 128, M / 128), gF(FF / 128, M / 128);
  dim3 gV(VV / 128, M / 128), gP(384 / 128, M / 128);

  // ---- encoder: x = LN(embed[ids]) ----
  slm_ln<<<M, 256, 0, stream>>>(token_ids, embed, nullptr, enc_g, enc_b, x);

  for (int l = 0; l < NL; ++l) {
    slm_ln<<<M, 256, 0, stream>>>(nullptr, nullptr, x, ln1_g + l * DIM, ln1_b + l * DIM, h);
    slm_gemm<<<gD, 256, 0, stream>>>(h, wqT + l * DD, qb, nullptr, nullptr, M, DIM, DIM);
    slm_gemm<<<gD, 256, 0, stream>>>(h, wkT + l * DD, kb, nullptr, nullptr, M, DIM, DIM);
    slm_gemm<<<gD, 256, 0, stream>>>(h, wvT + l * DD, vb, nullptr, nullptr, M, DIM, DIM);
    slm_rope<<<(BB * SS * NH * 32) / 256, 256, 0, stream>>>(qb);
    slm_rope<<<(BB * SS * NH * 32) / 256, 256, 0, stream>>>(kb);
    slm_attn<<<dim3(SS, NH, BB), 256, 0, stream>>>(qb, kb, vb, ob);
    slm_gemm<<<gD, 256, 0, stream>>>(ob, woT + l * DD, x, x, nullptr, M, DIM, DIM);  // x += o@Wo
    slm_ln<<<M, 256, 0, stream>>>(nullptr, nullptr, x, ln2_g + l * DIM, ln2_b + l * DIM, h);
    slm_gemm<<<gF, 256, 0, stream>>>(h, w1T + l * DF, hid, nullptr, nullptr, M, FF, DIM);
    slm_lif<<<(BB * FF) / 256, 256, 0, stream>>>(hid, spk);
    slm_gemm<<<gD, 256, 0, stream>>>(spk, w2T + l * DF, x, x, nullptr, M, DIM, FF);  // x += sp@W2
  }

  // ---- head: out = x@Wout + pw * gelu(x@Wp1)@Wp2 ----
  slm_gemm<<<gP, 256, 0, stream>>>(x, wp1T, p1, nullptr, nullptr, M, 384, DIM);
  slm_gelu<<<(M * 384) / 256, 256, 0, stream>>>(p1, M * 384);
  slm_gemm<<<gV, 256, 0, stream>>>(p1, wp2T, out, nullptr, nullptr, M, VV, 384);  // out = prior
  slm_gemm<<<gV, 256, 0, stream>>>(x, woutT, out, out, pw, M, VV, DIM);           // out = x@Wout + pw*out
}
